// Decoder_89240830476352
// MI455X (gfx1250) — matrix operand order repaired and checked by reading, not compiled, not run on hardware
//
#include <hip/hip_runtime.h>
#include <hip/hip_bf16.h>
#include <stdint.h>

#define B_       64
#define TENC_    512
#define E_       256
#define C_       256
#define K_       5
#define NL_      4
#define OUTD_    400
#define OUTDP_   416      // padded to multiple of 32 for WMMA K-loop
#define STEPS_   128
#define SQ05_    0.70710678118654752f
#define SQRTT_   22.627416997969522f   // sqrt(512)

typedef __attribute__((ext_vector_type(16))) _Float16 v16h;
typedef __attribute__((ext_vector_type(8)))  _Float16 v8h;
typedef __attribute__((ext_vector_type(8)))  float    v8f;

// ---------------------------------------------------------------------------
// Generic WMMA GEMM:  D[M=64][N] = act( A[64][Kin] * W[N][Kin]^T + bias[N] )
// A, W are f16 row-major (W is pre-transposed "out x in").  One wave per
// 16x16 tile, K swept 32 at a time with v_wmma_f32_16x16x32_f16.
// A/B fragments use contiguous k = k0+16g+s (same order in both); C/D layout per ISA (wave32).
// ---------------------------------------------------------------------------
__global__ __launch_bounds__(32)
void wmma_gemm_kernel(const _Float16* __restrict__ A, int lda,
                      const _Float16* __restrict__ W, int ldw,
                      const float* __restrict__ bias,
                      float* __restrict__ D, _Float16* __restrict__ Dh,
                      int N, int Kin, int act)
{
    const int nt = blockIdx.x;
    const int mt = blockIdx.y;
    const int l  = threadIdx.x;       // 0..31
    const int g  = l >> 4;            // half-wave (0/1)
    const int ln = l & 15;
    const int m  = mt * 16 + ln;      // A row held by this lane
    const int n  = nt * 16 + ln;      // W row (output column) held by this lane

    const _Float16* __restrict__ arow = A + (size_t)m * lda;
    const _Float16* __restrict__ wrow = W + (size_t)n * ldw;

    v8f c = {};
    for (int k0 = 0; k0 < Kin; k0 += 32) {
        // A 16x32 f16 fragment: lane (g) holds 16 contiguous K at k0+16g
        v16h af = *(const v16h*)(arow + k0 + g * 16);
        // B 32x16 f16 fragment: lane n holds 16 contiguous K at k0+16g
        v16h bf = *(const v16h*)(wrow + k0 + g * 16);
        c = __builtin_amdgcn_wmma_f32_16x16x32_f16(
                false, af, false, bf, (short)0, c, false, false);
    }
    const float bn = bias ? bias[n] : 0.0f;
#pragma unroll
    for (int r = 0; r < 8; ++r) {
        const int mr = mt * 16 + g * 8 + r;           // C/D layout: M = r + 8g
        float v = c[r] + bn;
        if (act == 1) v = v > 0.0f ? v : 0.0f;        // ReLU
        D[(size_t)mr * N + n] = v;
        if (Dh) Dh[(size_t)mr * N + n] = (_Float16)v;
    }
}

// ---------------------------------------------------------------------------
// Setup / data-movement kernels
// ---------------------------------------------------------------------------
__global__ void zero_u32_kernel(uint32_t* p, int n)
{
    int i = blockIdx.x * blockDim.x + threadIdx.x;
    if (i < n) p[i] = 0u;
}

__global__ void cvt_f16_kernel(const float* __restrict__ s, _Float16* __restrict__ d, int n)
{
    int i = blockIdx.x * blockDim.x + threadIdx.x;
    if (i < n) d[i] = (_Float16)s[i];
}

// pad columns with zeros: src rows x ic -> dst rows x oc
__global__ void cvt_pad_kernel(const float* __restrict__ s, _Float16* __restrict__ d,
                               int rows, int ic, int oc)
{
    int i = blockIdx.x * blockDim.x + threadIdx.x;
    if (i >= rows * oc) return;
    int r = i / oc, c = i % oc;
    d[i] = (c < ic) ? (_Float16)s[r * ic + c] : (_Float16)0.0f;
}

// (NL, 2C, C, K) -> (NL, 2C, K*C): Wt[i][o][k*C+c] = W[i][o][c][k]
__global__ void cvt_convw_kernel(const float* __restrict__ s, _Float16* __restrict__ d, int n)
{
    int i = blockIdx.x * blockDim.x + threadIdx.x;
    if (i >= n) return;
    int k = i % K_;
    int c = (i / K_) % C_;
    int o = (i / (K_ * C_)) % (2 * C_);
    int L = i / (K_ * C_ * 2 * C_);
    d[((size_t)(L * 2 * C_ + o)) * (K_ * C_) + k * C_ + c] = (_Float16)s[i];
}

// keys_pe = keys + sinusoid_embed(text_positions, E, 1.29)
__global__ void keyspe_kernel(const float* __restrict__ keys, const int* __restrict__ tpos,
                              float* __restrict__ kpe)
{
    int i = blockIdx.x * blockDim.x + threadIdx.x;
    if (i >= B_ * TENC_ * E_) return;
    int e = i % E_;
    int tp = (i / E_) % TENC_;
    int b = i / (E_ * TENC_);
    int pos = tpos[b * TENC_ + tp];
    float emb = 0.0f;
    if (pos != 0) {
        float denom = __powf(10000.0f, (float)(2 * (e / 2)) / (float)E_);
        float ang = 1.29f * (float)pos / denom;
        emb = (e & 1) ? __cosf(ang) : __sinf(ang);
    }
    kpe[i] = keys[i] + emb;
}

// shift conv ring buffer (B,K,C) up one row and append x
__global__ void shift_append_kernel(_Float16* __restrict__ buf, const _Float16* __restrict__ xh)
{
    int i = blockIdx.x * blockDim.x + threadIdx.x;   // B*C
    if (i >= B_ * C_) return;
    int b = i >> 8, c = i & 255;
    _Float16* p = buf + (size_t)b * (K_ * C_) + c;
    _Float16 r1 = p[1 * C_], r2 = p[2 * C_], r3 = p[3 * C_], r4 = p[4 * C_];
    p[0 * C_] = r1; p[1 * C_] = r2; p[2 * C_] = r3; p[3 * C_] = r4;
    p[4 * C_] = xh[i];
}

// pre-net GLU with residual: x = (a*sigmoid(g) + x) * sqrt(0.5)
__global__ void glu_res_kernel(const float* __restrict__ y,
                               float* __restrict__ x, _Float16* __restrict__ xh)
{
    int i = blockIdx.x * blockDim.x + threadIdx.x;
    if (i >= B_ * C_) return;
    int b = i >> 8, c = i & 255;
    float a = y[b * (2 * C_) + c];
    float gg = y[b * (2 * C_) + C_ + c];
    float v = (a * (1.0f / (1.0f + __expf(-gg))) + x[i]) * SQ05_;
    x[i] = v; xh[i] = (_Float16)v;
}

// attention-layer GLU (no residual) + query positional embed; save res = old x
__global__ void glu_fpos_kernel(const float* __restrict__ y,
                                float* __restrict__ x, _Float16* __restrict__ xh,
                                float* __restrict__ res, int t)
{
    int i = blockIdx.x * blockDim.x + threadIdx.x;
    if (i >= B_ * C_) return;
    int b = i >> 8, c = i & 255;
    res[i] = x[i];
    float a = y[b * (2 * C_) + c];
    float gg = y[b * (2 * C_) + C_ + c];
    float denom = __powf(10000.0f, (float)(2 * (c / 2)) / (float)C_);
    float ang = (float)(t + 1) / denom;
    float fpos = (c & 1) ? __cosf(ang) : __sinf(ang);
    float v = a * (1.0f / (1.0f + __expf(-gg))) + fpos;
    x[i] = v; xh[i] = (_Float16)v;
}

// windowed attention: scores only at pos in [la-1, la+3) ∩ [0,TENC); the
// -1e9 mask underflows to exact 0 in the reference softmax, so this is exact.
__global__ __launch_bounds__(256)
void attend_kernel(const float* __restrict__ q, const float* __restrict__ kpe,
                   const float* __restrict__ vals, const int* __restrict__ labuf,
                   int layer, _Float16* __restrict__ ctxh,
                   float* __restrict__ align_out, int t)
{
    const int b = blockIdx.x;
    const int e = threadIdx.x;          // 0..255
    const int la = labuf[layer];
    const int p0 = la - 1;

    float qv = q[b * E_ + e];
    float part[4];
#pragma unroll
    for (int j = 0; j < 4; ++j) {
        int p = p0 + j;
        part[j] = (p >= 0 && p < TENC_)
                    ? qv * kpe[((size_t)b * TENC_ + p) * E_ + e] : 0.0f;
    }
    // wave32 reduction
#pragma unroll
    for (int j = 0; j < 4; ++j)
        for (int off = 16; off > 0; off >>= 1)
            part[j] += __shfl_down(part[j], off, 32);

    __shared__ float sred[8][4];
    __shared__ float wgt[4];
    int wv = e >> 5, lane = e & 31;
    if (lane == 0) {
#pragma unroll
        for (int j = 0; j < 4; ++j) sred[wv][j] = part[j];
    }
    __syncthreads();
    if (e == 0) {
        float s[4];
#pragma unroll
        for (int j = 0; j < 4; ++j) {
            float a = 0.0f;
            for (int w8 = 0; w8 < 8; ++w8) a += sred[w8][j];
            int p = p0 + j;
            s[j] = (p >= 0 && p < TENC_) ? a : -1e30f;
        }
        float mx = s[0];
#pragma unroll
        for (int j = 1; j < 4; ++j) mx = s[j] > mx ? s[j] : mx;
        float ex[4], sum = 0.0f;
#pragma unroll
        for (int j = 0; j < 4; ++j) {
            ex[j] = (s[j] <= -1e29f) ? 0.0f : __expf(s[j] - mx);
            sum += ex[j];
        }
#pragma unroll
        for (int j = 0; j < 4; ++j) wgt[j] = ex[j] / sum;
    }
    __syncthreads();

    float ctx = 0.0f;
#pragma unroll
    for (int j = 0; j < 4; ++j) {
        int p = p0 + j;
        if (p >= 0 && p < TENC_)
            ctx += wgt[j] * vals[((size_t)b * TENC_ + p) * E_ + e];
    }
    ctx *= SQRTT_;
    ctxh[b * E_ + e] = (_Float16)ctx;

    // alignments output = 2 * layer-0 attention (reference's ave=ave+ave bug)
    if (align_out && e < 4) {
        int p = p0 + e;
        if (p >= 0 && p < TENC_)
            align_out[((size_t)b * STEPS_ + t) * TENC_ + p] = 2.0f * wgt[e];
    }
}

// force_monotonic: la <- argmax over batch-0 window (first index on ties)
__global__ __launch_bounds__(32)
void argmax_kernel(const float* __restrict__ q, const float* __restrict__ kpe,
                   int* __restrict__ labuf, int layer)
{
    int l = threadIdx.x;
    int la = labuf[layer], p0 = la - 1;
    __shared__ float s[4];
    if (l < 4) {
        int p = p0 + l;
        float acc = -1e30f;
        if (p >= 0 && p < TENC_) {
            acc = 0.0f;
            const float* kr = kpe + (size_t)p * E_;   // batch 0
            for (int e = 0; e < E_; ++e) acc += q[e] * kr[e];
        }
        s[l] = acc;
    }
    __syncthreads();
    if (l == 0) {
        int best = 0; float bv = s[0];
        for (int j = 1; j < 4; ++j) if (s[j] > bv) { bv = s[j]; best = j; }
        labuf[layer] = p0 + best;
    }
}

// x = ((o + x_attn)*sqrt(.5) + res)*sqrt(.5); optionally emit decoder_states
__global__ void combine_kernel(const float* __restrict__ o, const float* __restrict__ res,
                               float* __restrict__ x, _Float16* __restrict__ xh,
                               float* __restrict__ states_out, int t)
{
    int i = blockIdx.x * blockDim.x + threadIdx.x;
    if (i >= B_ * C_) return;
    int b = i >> 8, c = i & 255;
    float v = (o[i] + x[i]) * SQ05_;
    v = (v + res[i]) * SQ05_;
    x[i] = v; xh[i] = (_Float16)v;
    if (states_out) states_out[((size_t)b * STEPS_ + t) * C_ + c] = v;
}

// sigmoid output, next-step input, done gate (400-dot with W_fc)
__global__ __launch_bounds__(256)
void finalize_kernel(const float* __restrict__ ylast, const float* __restrict__ wfc,
                     const float* __restrict__ bfc, float* __restrict__ out_base,
                     float* __restrict__ done_base, _Float16* __restrict__ curh, int t)
{
    int b = blockIdx.x, tid = threadIdx.x;
    __shared__ float red[256];
    float acc = 0.0f;
    for (int o = tid; o < OUTD_; o += 256) {
        float yv = ylast[b * OUTD_ + o];
        float ov = 1.0f / (1.0f + __expf(-yv));
        out_base[((size_t)b * STEPS_ + t) * OUTD_ + o] = ov;
        curh[b * OUTDP_ + o] = (_Float16)ov;
        acc += yv * wfc[o];
    }
    red[tid] = acc;
    __syncthreads();
    for (int s2 = 128; s2 > 0; s2 >>= 1) {
        if (tid < s2) red[tid] += red[tid + s2];
        __syncthreads();
    }
    if (tid == 0)
        done_base[(size_t)b * STEPS_ + t] = 1.0f / (1.0f + __expf(-(red[0] + bfc[0])));
}

// ---------------------------------------------------------------------------
// host
// ---------------------------------------------------------------------------
extern "C" void kernel_launch(void* const* d_in, const int* in_sizes, int n_in,
                              void* d_out, int out_size, void* d_ws, size_t ws_size,
                              hipStream_t stream)
{
    (void)in_sizes; (void)n_in; (void)out_size; (void)ws_size;

    const float* keys    = (const float*)d_in[0];
    const float* values  = (const float*)d_in[1];
    const int*   tpos    = (const int*)d_in[2];
    const float* W_pre1  = (const float*)d_in[4];
    const float* b_pre1  = (const float*)d_in[5];
    const float* Wg_pre  = (const float*)d_in[6];
    const float* bg_pre  = (const float*)d_in[7];
    const float* Wg_conv = (const float*)d_in[8];
    const float* bg_conv = (const float*)d_in[9];
    const float* Wq      = (const float*)d_in[10];
    const float* bq      = (const float*)d_in[11];
    const float* Wo      = (const float*)d_in[12];
    const float* bo      = (const float*)d_in[13];
    const float* W_last  = (const float*)d_in[14];
    const float* b_last  = (const float*)d_in[15];
    const float* W_fc    = (const float*)d_in[16];
    const float* b_fc    = (const float*)d_in[17];

    // output regions (flat fp32, reference tuple order)
    float* out_base   = (float*)d_out;                               // (B,T,400)
    float* align_base = out_base + (size_t)B_ * STEPS_ * OUTD_;      // (B,T,512)
    float* done_base  = align_base + (size_t)B_ * STEPS_ * TENC_;    // (B,T,1)
    float* state_base = done_base + (size_t)B_ * STEPS_;             // (B,T,256)

    // workspace carve (bump allocator, 256B aligned)
    uint8_t* wsp = (uint8_t*)d_ws;
    auto carve = [&](size_t bytes) -> void* {
        void* p = (void*)wsp;
        wsp += (bytes + 255) & ~(size_t)255;
        return p;
    };
    float*    kpe    = (float*)   carve((size_t)B_ * TENC_ * E_ * 4);
    _Float16* Wpre1h = (_Float16*)carve((size_t)C_ * OUTDP_ * 2);
    _Float16* WgPreh = (_Float16*)carve((size_t)NL_ * 2 * C_ * K_ * C_ * 2);
    _Float16* WgCnvh = (_Float16*)carve((size_t)NL_ * 2 * C_ * K_ * C_ * 2);
    _Float16* Wqh    = (_Float16*)carve((size_t)NL_ * E_ * C_ * 2);
    _Float16* Woh    = (_Float16*)carve((size_t)NL_ * C_ * E_ * 2);
    _Float16* Wlasth = (_Float16*)carve((size_t)OUTD_ * C_ * 2);
    _Float16* curh   = (_Float16*)carve((size_t)B_ * OUTDP_ * 2);
    _Float16* prebuf = (_Float16*)carve((size_t)NL_ * B_ * K_ * C_ * 2);
    _Float16* cnvbuf = (_Float16*)carve((size_t)NL_ * B_ * K_ * C_ * 2);
    float*    xbuf   = (float*)   carve((size_t)B_ * C_ * 4);
    _Float16* xh     = (_Float16*)carve((size_t)B_ * C_ * 2);
    float*    ybuf   = (float*)   carve((size_t)B_ * 2 * C_ * 4);
    float*    qbuf   = (float*)   carve((size_t)B_ * E_ * 4);
    float*    resbuf = (float*)   carve((size_t)B_ * C_ * 4);
    _Float16* ctxh   = (_Float16*)carve((size_t)B_ * E_ * 2);
    float*    obuf   = (float*)   carve((size_t)B_ * C_ * 4);
    float*    ylast  = (float*)   carve((size_t)B_ * OUTD_ * 4);
    int*      labuf  = (int*)     carve(256);

    // ---- per-launch setup (deterministic re-init of all state) ----
    {
        int n;
        n = B_ * TENC_ * E_;
        keyspe_kernel<<<(n + 255) / 256, 256, 0, stream>>>(keys, tpos, kpe);

        n = C_ * OUTDP_;
        cvt_pad_kernel<<<(n + 255) / 256, 256, 0, stream>>>(W_pre1, Wpre1h, C_, OUTD_, OUTDP_);

        n = NL_ * 2 * C_ * C_ * K_;
        cvt_convw_kernel<<<(n + 255) / 256, 256, 0, stream>>>(Wg_pre, WgPreh, n);
        cvt_convw_kernel<<<(n + 255) / 256, 256, 0, stream>>>(Wg_conv, WgCnvh, n);

        n = NL_ * E_ * C_;
        cvt_f16_kernel<<<(n + 255) / 256, 256, 0, stream>>>(Wq, Wqh, n);
        cvt_f16_kernel<<<(n + 255) / 256, 256, 0, stream>>>(Wo, Woh, n);

        n = OUTD_ * C_;
        cvt_f16_kernel<<<(n + 255) / 256, 256, 0, stream>>>(W_last, Wlasth, n);

        n = (B_ * OUTDP_) / 2;      // curh zeros (u32 count)
        zero_u32_kernel<<<(n + 255) / 256, 256, 0, stream>>>((uint32_t*)curh, n);
        n = (NL_ * B_ * K_ * C_) / 2;
        zero_u32_kernel<<<(n + 255) / 256, 256, 0, stream>>>((uint32_t*)prebuf, n);
        zero_u32_kernel<<<(n + 255) / 256, 256, 0, stream>>>((uint32_t*)cnvbuf, n);
        zero_u32_kernel<<<1, 256, 0, stream>>>((uint32_t*)labuf, NL_);
        n = B_ * STEPS_ * TENC_;    // alignments region must be zero-filled
        zero_u32_kernel<<<(n + 255) / 256, 256, 0, stream>>>((uint32_t*)align_base, n);
    }

    auto gemm = [&](const _Float16* A, int lda, const _Float16* W, int ldw,
                    const float* bias, float* D, _Float16* Dh, int N, int Kin, int act) {
        dim3 grid(N / 16, B_ / 16);
        wmma_gemm_kernel<<<grid, 32, 0, stream>>>(A, lda, W, ldw, bias, D, Dh, N, Kin, act);
    };

    // ---- 128 sequential decode steps ----
    for (int t = 0; t < STEPS_; ++t) {
        // x = relu(cur @ W_pre1^T + b)   (64 x 256, K=416 padded)
        gemm(curh, OUTDP_, Wpre1h, OUTDP_, b_pre1, xbuf, xh, C_, OUTDP_, 1);

        // pre-net causal Conv1dGLU stack (residual)
        for (int i = 0; i < NL_; ++i) {
            _Float16* buf = prebuf + (size_t)i * B_ * K_ * C_;
            shift_append_kernel<<<(B_ * C_) / 256, 256, 0, stream>>>(buf, xh);
            gemm(buf, K_ * C_, WgPreh + (size_t)i * 2 * C_ * K_ * C_, K_ * C_,
                 bg_pre + i * 2 * C_, ybuf, nullptr, 2 * C_, K_ * C_, 0);
            glu_res_kernel<<<(B_ * C_) / 256, 256, 0, stream>>>(ybuf, xbuf, xh);
        }

        // Conv1dGLU (no residual) + windowed attention
        for (int i = 0; i < NL_; ++i) {
            _Float16* buf = cnvbuf + (size_t)i * B_ * K_ * C_;
            shift_append_kernel<<<(B_ * C_) / 256, 256, 0, stream>>>(buf, xh);
            gemm(buf, K_ * C_, WgCnvh + (size_t)i * 2 * C_ * K_ * C_, K_ * C_,
                 bg_conv + i * 2 * C_, ybuf, nullptr, 2 * C_, K_ * C_, 0);
            glu_fpos_kernel<<<(B_ * C_) / 256, 256, 0, stream>>>(ybuf, xbuf, xh, resbuf, t);
            // q = x @ Wq^T + bq
            gemm(xh, C_, Wqh + (size_t)i * E_ * C_, C_, bq + i * E_,
                 qbuf, nullptr, E_, C_, 0);
            attend_kernel<<<B_, 256, 0, stream>>>(qbuf, kpe, values, labuf, i, ctxh,
                                                  (i == 0) ? align_base : nullptr, t);
            argmax_kernel<<<1, 32, 0, stream>>>(qbuf, kpe, labuf, i);
            // out = ctx @ Wo^T + bo
            gemm(ctxh, E_, Woh + (size_t)i * C_ * E_, E_, bo + i * C_,
                 obuf, nullptr, C_, E_, 0);
            combine_kernel<<<(B_ * C_) / 256, 256, 0, stream>>>(
                obuf, resbuf, xbuf, xh, (i == NL_ - 1) ? state_base : nullptr, t);
        }

        // y = x @ W_last^T + b; out = sigmoid(y); done = sigmoid(y . W_fc + b_fc)
        gemm(xh, C_, Wlasth, C_, b_last, ylast, nullptr, OUTD_, C_, 0);
        finalize_kernel<<<B_, 256, 0, stream>>>(ylast, W_fc, b_fc,
                                                out_base, done_base, curh, t);
    }
}
